// TimeModel_72043781423300
// MI455X (gfx1250) — matrix operand order repaired and checked by reading, not compiled, not run on hardware
//
#include <hip/hip_runtime.h>
#include <hip/hip_bf16.h>
#include <math.h>

// ---------------------------------------------------------------------------
// TimesNet-style TimeModel for MI455X (gfx1250).
//   * Inception(1,3,5)/3 folded into a single 5x5 conv -> implicit-GEMM WMMA
//   * On-device DFT amplitude spectrum as f16 WMMA GEMMs vs Fourier matrices
//   * Data-dependent periods kept in a device descriptor (graph-capture safe)
// All heavy math runs through v_wmma_f32_16x16x32_f16 (f32 accumulation).
// ---------------------------------------------------------------------------

typedef __attribute__((ext_vector_type(16))) _Float16 v16h;
typedef __attribute__((ext_vector_type(8)))  _Float16 v8h;
typedef __attribute__((ext_vector_type(8)))  float    v8f;

#define T_LEN   365
#define DM      16
#define KPER    5
#define BATCH   512
#define NCH     4
#define TPAD    384      // 12 * 32 (DFT K padding)
#define NKCHUNK 12
#define NFTILE  23       // ceil(365/16)
#define LMAX    560      // > max padded length (546, at p=182)
#define NPAIR   13       // ceil(25 taps / 2)

#define NA_DFT  (2*NFTILE*NKCHUNK*32*16)
#define NA_CONV (2*NPAIR*32*16)

// desc ints: [0..4]=idx  [5..9]=p  [10..14]=rows  [15..19]=len
#define DESC_IDX 0
#define DESC_P   5
#define DESC_R   10
#define DESC_L   15

static __device__ inline v8f wmma16(v16h a, v16h b, v8f c) {
  return __builtin_amdgcn_wmma_f32_16x16x32_f16(false, a, false, b, (short)0, c,
                                                false, false);
}

// A-operand (16x32 f16) K order used here (B matches): lane L holds row m=L%16; element e of
// the per-lane v16h maps to K = kh+e with kh = (L>=16)*16.
static __device__ inline int kmapA(int lane, int e) {
  int kh = (lane >= 16) ? 16 : 0;
  return kh + e;
}

// ---------------------------------------------------------------------------
// Pack kernel: Fourier A-tiles (cos/sin), combined-conv A-tiles, combined bias
// ---------------------------------------------------------------------------
__global__ void pack_kernel(const float* w10, const float* b10, const float* w11,
                            const float* b11, const float* w12, const float* b12,
                            const float* w20, const float* b20, const float* w21,
                            const float* b21, const float* w22, const float* b22,
                            _Float16* apackDft, _Float16* apackConv,
                            float* bc1, float* bc2) {
  int i = blockIdx.x * blockDim.x + threadIdx.x;
  if (i < NA_DFT) {
    int e = i & 15, lane = (i >> 4) & 31;
    int r = i >> 9;
    int chunk = r % NKCHUNK; r /= NKCHUNK;
    int ft = r % NFTILE;
    int trig = r / NFTILE;
    int K = kmapA(lane, e);
    int t = chunk * 32 + K;
    int f = ft * 16 + (lane & 15);
    float v = 0.f;
    if (t < T_LEN && f < T_LEN) {
      int m = (int)(((long long)t * f) % T_LEN);
      float ang = (float)m * (6.283185307179586f / (float)T_LEN);
      v = trig ? sinf(ang) : cosf(ang);   // |X| uses Im^2, sign irrelevant
    }
    apackDft[i] = (_Float16)v;
    return;
  }
  int j = i - NA_DFT;
  if (j < NA_CONV) {
    int e = j & 15, lane = (j >> 4) & 31;
    int r = j >> 9;
    int pair = r % NPAIR;
    int layer = r / NPAIR;
    int K = kmapA(lane, e);
    int tap = pair * 2 + (K >> 4);
    int ci = K & 15;
    int co = lane & 15;
    float v = 0.f;
    if (tap < 25) {
      int dy = tap / 5, dx = tap % 5;
      const float* w5 = layer ? w22 : w12;
      const float* w3 = layer ? w21 : w11;
      const float* w1 = layer ? w20 : w10;
      float s = w5[((co * DM + ci) * 5 + dy) * 5 + dx];
      if (dy >= 1 && dy <= 3 && dx >= 1 && dx <= 3)
        s += w3[((co * DM + ci) * 3 + (dy - 1)) * 3 + (dx - 1)];
      if (dy == 2 && dx == 2) s += w1[co * DM + ci];
      v = s * (1.f / 3.f);
    }
    apackConv[j] = (_Float16)v;
    return;
  }
  int k = j - NA_CONV;
  if (k < 32) {
    int layer = k >> 4, co = k & 15;
    const float* x0 = layer ? b20 : b10;
    const float* x1 = layer ? b21 : b11;
    const float* x2 = layer ? b22 : b12;
    float* dst = layer ? bc2 : bc1;
    dst[co] = (x0[co] + x1[co] + x2[co]) * (1.f / 3.f);
  }
}

// ---------------------------------------------------------------------------
// Embedding: circular conv1d(k=3) + positional encoding.
// Writes residual directly into d_out channel slice, plus f16 copies
// (emb_h: [B][T][16] for conv B-operands; embT_h: [B][16][TPAD] for DFT).
// ---------------------------------------------------------------------------
__global__ void embed_kernel(const float* x, const float* tokw, int c,
                             float* dout, _Float16* emb_h, _Float16* embT_h) {
  int b = blockIdx.x;
  int t = threadIdx.x;              // block = TPAD threads
  if (t >= TPAD) return;
  if (t >= T_LEN) {                 // zero K-padding for the DFT GEMM
    for (int d = 0; d < DM; ++d) embT_h[(b * DM + d) * TPAD + t] = (_Float16)0.f;
    return;
  }
  int tm = (t + T_LEN - 1) % T_LEN;
  int tp = (t + 1) % T_LEN;
  float xm = x[(b * T_LEN + tm) * NCH + c];
  float x0 = x[(b * T_LEN + t) * NCH + c];
  float xp = x[(b * T_LEN + tp) * NCH + c];
  for (int d = 0; d < DM; ++d) {
    float v = tokw[d * 3 + 0] * xm + tokw[d * 3 + 1] * x0 + tokw[d * 3 + 2] * xp;
    float divv = expf(-0.5756462732485114f * (float)(2 * (d >> 1)));  // ln(1e4)/16
    v += (d & 1) ? cosf((float)t * divv) : sinf((float)t * divv);
    dout[(b * T_LEN + t) * 64 + c * DM + d] = v;     // residual term
    emb_h[(b * T_LEN + t) * DM + d] = (_Float16)v;
    embT_h[(b * DM + d) * TPAD + t] = (_Float16)v;
  }
}

// ---------------------------------------------------------------------------
// DFT amplitudes via WMMA: per (b, f-tile) wave computes Re/Im[16f x 16d]
// accumulating over 12 K=32 chunks, then mean_d |X| -> ampD[b][f].
// ---------------------------------------------------------------------------
__global__ void __launch_bounds__(32)
dft_amp_kernel(const _Float16* embT_h, const _Float16* apackDft, float* ampD) {
  int b = blockIdx.x, ft = blockIdx.y, lane = threadIdx.x;
  v8f re = {}, im = {};
  const v16h* Ac = (const v16h*)apackDft;
  const v16h* As = Ac + NFTILE * NKCHUNK * 32;
  int d = lane & 15;
  int koff = (lane >= 16) ? 16 : 0;   // B layout: lanes>=16 carry K upper half
  const _Float16* brow = embT_h + (b * DM + d) * TPAD + koff;
  for (int chunk = 0; chunk < NKCHUNK; ++chunk) {
    v16h bv = *(const v16h*)(brow + chunk * 32);
    v16h ac = Ac[(ft * NKCHUNK + chunk) * 32 + lane];
    v16h as = As[(ft * NKCHUNK + chunk) * 32 + lane];
    re = wmma16(ac, bv, re);
    im = wmma16(as, bv, im);
  }
  __shared__ float s[16][16];
  int frow = (lane < 16) ? 0 : 8;     // D layout: VGPR j -> M = j (+8 upper half)
  for (int j = 0; j < 8; ++j)
    s[frow + j][d] = sqrtf(re[j] * re[j] + im[j] * im[j]);
  __syncthreads();
  if (lane < 16) {
    float sum = 0.f;
    for (int q = 0; q < 16; ++q) sum += s[lane][q];
    int f = ft * 16 + lane;
    if (f < T_LEN) ampD[b * T_LEN + f] = sum * (1.f / 16.f);
  }
}

__global__ void freq_kernel(const float* ampD, float* freq) {
  int f = blockIdx.x;
  int tid = threadIdx.x;
  __shared__ float sh[256];
  float s = 0.f;
  for (int b = tid; b < BATCH; b += 256) s += ampD[b * T_LEN + f];
  sh[tid] = s;
  __syncthreads();
  for (int st = 128; st > 0; st >>= 1) {
    if (tid < st) sh[tid] += sh[tid + st];
    __syncthreads();
  }
  if (tid == 0) freq[f] = (f == 0) ? 0.f : sh[0] * (1.f / BATCH);
}

__global__ void topk_kernel(const float* freq, int* desc) {
  if (threadIdx.x != 0) return;
  int chosen[KPER];
  for (int k = 0; k < KPER; ++k) {
    float bv = -1.f;
    int bi = 1;
    for (int f = 1; f < T_LEN; ++f) {
      bool skip = false;
      for (int q = 0; q < k; ++q)
        if (chosen[q] == f) skip = true;
      if (!skip && freq[f] > bv) { bv = freq[f]; bi = f; }
    }
    chosen[k] = bi;
    int p = T_LEN / bi;
    int rows = (T_LEN + p - 1) / p;
    desc[DESC_IDX + k] = bi;
    desc[DESC_P + k] = p;
    desc[DESC_R + k] = rows;
    desc[DESC_L + k] = rows * p;
  }
}

__global__ void wsoft_kernel(const float* ampD, const int* desc, float* wsoft) {
  int b = blockIdx.x * blockDim.x + threadIdx.x;
  if (b >= BATCH) return;
  float w[KPER];
  float m = -1e30f;
  for (int k = 0; k < KPER; ++k) {
    w[k] = ampD[b * T_LEN + desc[DESC_IDX + k]];
    m = fmaxf(m, w[k]);
  }
  float sum = 0.f;
  for (int k = 0; k < KPER; ++k) { w[k] = expf(w[k] - m); sum += w[k]; }
  float inv = 1.f / sum;
  for (int k = 0; k < KPER; ++k) wsoft[b * KPER + k] = w[k] * inv;
}

// ---------------------------------------------------------------------------
// Combined 5x5 conv as implicit-GEMM WMMA over a rows x p grid (period from
// device descriptor). One wave per 16-position output tile; 13 WMMAs cover
// 25 taps (taps paired into the two K=16 halves: lanes<16 even tap, lanes>=16
// odd tap, each lane loads one contiguous 16-channel input row from L2).
//   phase 0: emb -> conv1 -> +bias -> exact GELU -> mid (f16), all len pos
//   phase 1: mid -> conv2 -> +bias -> * wsoft[b][per], += d_out, t < 365
// ---------------------------------------------------------------------------
__global__ void __launch_bounds__(128)
conv_kernel(const _Float16* src, const _Float16* apackConv, const float* bias,
            const int* desc, int per, int phase, _Float16* mid, float* dout,
            const float* wsoft, int c) {
  int b = blockIdx.x;
  int wave = threadIdx.x >> 5;
  int lane = threadIdx.x & 31;
  int p = desc[DESC_P + per];
  int rows = desc[DESC_R + per];
  int len = desc[DESC_L + per];
  int limit = phase ? T_LEN : len;
  int tile = blockIdx.y * 4 + wave;
  int t0 = tile * 16;
  if (t0 >= limit) return;            // wave-uniform: EXEC stays all-ones
  int n = lane & 15;
  int tcur = t0 + n;
  int y = tcur / p;
  int x = tcur - y * p;
  int tapAdd = (lane >= 16) ? 1 : 0;
  v8f acc = {};
  const v16h* A = (const v16h*)apackConv + (phase ? NPAIR * 32 : 0);
  for (int pair = 0; pair < NPAIR; ++pair) {
    int tap = pair * 2 + tapAdd;
    v16h bv = {};
    if (tap < 25) {
      int ys = y + tap / 5 - 2;
      int xs = x + tap % 5 - 2;
      if (ys >= 0 && ys < rows && xs >= 0 && xs < p) {
        int ts = ys * p + xs;
        if (phase) {
          bv = *(const v16h*)(src + (b * LMAX + ts) * DM);
        } else if (ts < T_LEN) {      // zero-pad region of the reshape
          bv = *(const v16h*)(src + (b * T_LEN + ts) * DM);
        }
      }
    }
    v16h av = A[pair * 32 + lane];
    acc = wmma16(av, bv, acc);        // v_wmma_f32_16x16x32_f16
  }
  if (tcur >= limit) return;          // mask partial tile after the WMMAs
  int cobase = (lane < 16) ? 0 : 8;   // D layout: VGPR j -> co = cobase + j
  if (!phase) {
    v8h outv;
    for (int j = 0; j < 8; ++j) {
      float v = acc[j] + bias[cobase + j];
      v = 0.5f * v * (1.f + erff(v * 0.70710678118f));  // exact GELU
      outv[j] = (_Float16)v;
    }
    *(v8h*)(mid + (b * LMAX + tcur) * DM + cobase) = outv;
  } else {
    float w = wsoft[b * KPER + per];
    float* dp = dout + (b * T_LEN + tcur) * 64 + c * DM + cobase;
    for (int j = 0; j < 8; ++j) dp[j] += w * (acc[j] + bias[cobase + j]);
  }
}

// ---------------------------------------------------------------------------
extern "C" void kernel_launch(void* const* d_in, const int* in_sizes, int n_in,
                              void* d_out, int out_size, void* d_ws,
                              size_t ws_size, hipStream_t stream) {
  (void)in_sizes; (void)n_in; (void)out_size; (void)ws_size;
  const float* x    = (const float*)d_in[0];
  const float* tokw = (const float*)d_in[1];
  const float* c1w0 = (const float*)d_in[2],  *c1b0 = (const float*)d_in[3];
  const float* c1w1 = (const float*)d_in[4],  *c1b1 = (const float*)d_in[5];
  const float* c1w2 = (const float*)d_in[6],  *c1b2 = (const float*)d_in[7];
  const float* c2w0 = (const float*)d_in[8],  *c2b0 = (const float*)d_in[9];
  const float* c2w1 = (const float*)d_in[10], *c2b1 = (const float*)d_in[11];
  const float* c2w2 = (const float*)d_in[12], *c2b2 = (const float*)d_in[13];
  float* out = (float*)d_out;

  char* ws = (char*)d_ws;
  size_t off = 0;
  auto take = [&](size_t bytes) {
    size_t o = off;
    off += (bytes + 255) & ~(size_t)255;
    return o;
  };
  float*    bc1       = (float*)(ws + take(16 * 4));
  float*    bc2       = (float*)(ws + take(16 * 4));
  int*      desc      = (int*)(ws + take(20 * 4));
  float*    freq      = (float*)(ws + take(T_LEN * 4));
  float*    wsoft     = (float*)(ws + take((size_t)BATCH * KPER * 4));
  _Float16* apackDft  = (_Float16*)(ws + take((size_t)NA_DFT * 2));
  _Float16* apackConv = (_Float16*)(ws + take((size_t)NA_CONV * 2));
  float*    ampD      = (float*)(ws + take((size_t)BATCH * T_LEN * 4));
  _Float16* emb_h     = (_Float16*)(ws + take((size_t)BATCH * T_LEN * DM * 2));
  _Float16* embT_h    = (_Float16*)(ws + take((size_t)BATCH * DM * TPAD * 2));
  _Float16* mid_h     = (_Float16*)(ws + take((size_t)BATCH * LMAX * DM * 2));

  int packN = NA_DFT + NA_CONV + 32;
  pack_kernel<<<(packN + 255) / 256, 256, 0, stream>>>(
      c1w0, c1b0, c1w1, c1b1, c1w2, c1b2, c2w0, c2b0, c2w1, c2b1, c2w2, c2b2,
      apackDft, apackConv, bc1, bc2);

  for (int c = 0; c < NCH; ++c) {
    embed_kernel<<<BATCH, TPAD, 0, stream>>>(x, tokw, c, out, emb_h, embT_h);
    dft_amp_kernel<<<dim3(BATCH, NFTILE), 32, 0, stream>>>(embT_h, apackDft, ampD);
    freq_kernel<<<T_LEN, 256, 0, stream>>>(ampD, freq);
    topk_kernel<<<1, 32, 0, stream>>>(freq, desc);
    wsoft_kernel<<<(BATCH + 255) / 256, 256, 0, stream>>>(ampD, desc, wsoft);
    for (int per = 0; per < KPER; ++per) {
      // phase 0: tiles cover LMAX (worst-case padded length, device-bounded)
      conv_kernel<<<dim3(BATCH, 9), 128, 0, stream>>>(
          emb_h, apackConv, bc1, desc, per, 0, mid_h, out, wsoft, c);
      // phase 1: only t < 365 outputs needed
      conv_kernel<<<dim3(BATCH, 6), 128, 0, stream>>>(
          mid_h, apackConv, bc2, desc, per, 1, mid_h, out, wsoft, c);
    }
  }
}
